// GraphBlock_84061099917988
// MI455X (gfx1250) — matrix operand order repaired and checked by reading, not compiled, not run on hardware
//
#include <hip/hip_runtime.h>
#include <hip/hip_bf16.h>
#include <math.h>

typedef __bf16 v16bf __attribute__((ext_vector_type(16)));
typedef float  v8f   __attribute__((ext_vector_type(8)));

#define KI 64
#define NEG_SLOPE 0.2f

// ---------------------------------------------------------------- helpers
__device__ __forceinline__ float gelu_tanh(float v) {
    // jax.nn.gelu default (approximate=True)
    float c = 0.7978845608028654f * (v + 0.044715f * v * v * v);
    return 0.5f * v * (1.0f + tanhf(c));
}

__device__ __forceinline__ void atomicMaxF(float* addr, float val) {
    int* ai = (int*)addr;
    int old = __float_as_int(*addr);
    while (__int_as_float(old) < val) {
        int assumed = old;
        old = atomicCAS(ai, assumed, __float_as_int(val));
        if (old == assumed) break;
    }
}

// ---------------------------------------------------------------- fill
__global__ void k_fill(float* p, float v, int n) {
    int i = blockIdx.x * blockDim.x + threadIdx.x;
    if (i < n) p[i] = v;
}

// ---------------------------------------------------------------- GEMM (bf16 WMMA)
// Y[n_rows x Ko] = X[n_rows x 64] @ W[64 x Ko], fp32 in/out, bf16 multiply.
// Block: 128 threads (4 waves), one 16-row M tile per block.
__global__ __launch_bounds__(128)
void k_gemm_bf16(const float* __restrict__ X, const float* __restrict__ W,
                 float* __restrict__ Y, int n_rows, int Ko)
{
    __shared__ __bf16 Alds[16][KI + 2];    // +2 pad: break 128B-stride bank conflicts
    __shared__ __bf16 Wlds[128][KI + 2];   // W transposed: [col][k]

    const int t = threadIdx.x;
    const int mBase = blockIdx.x * 16;
    const bool fullTile = (mBase + 16 <= n_rows);

    for (int idx = t; idx < 16 * KI; idx += 128) {
        int r = idx >> 6, k = idx & 63;
        int row = mBase + r;
        float v = (row < n_rows) ? X[row * KI + k] : 0.0f;
        Alds[r][k] = (__bf16)v;
    }
    for (int idx = t; idx < Ko * KI; idx += 128) {
        int j = idx >> 6, k = idx & 63;
        Wlds[j][k] = (__bf16)W[k * Ko + j];
    }
    __syncthreads();

    const int lane = t & 31;
    const int wave = t >> 5;
    const int m    = lane & 15;   // A: M row / B: N col / D: N col
    const int g    = lane >> 4;

    const int nct = Ko >> 4;
    for (int ct = wave; ct < nct; ct += 4) {
        const int colBase = ct << 4;
        v8f acc = {};
        for (int ks = 0; ks < KI; ks += 32) {
            union { v16bf v; unsigned int u[8]; } af, bf;
            #pragma unroll
            for (int vv = 0; vv < 8; ++vv) {
                // ISA 16-bit A 16x32 layout: V0..3 -> K=2v+1:2v (+8g), V4..7 -> K=16+.. (+8g)
                int ka = ((vv < 4) ? (2 * vv + 8 * g) : (16 + 2 * (vv - 4) + 8 * g)) + ks;
                af.u[vv] = *(const unsigned int*)&Alds[m][ka];
                // B 32x16 layout matching A: V0..3 -> K=2v+1:2v (+8g), V4..7 -> K=16+.. (+8g), N = lane%16
                int kb = ((vv < 4) ? (2 * vv + 8 * g) : (16 + 2 * (vv - 4) + 8 * g)) + ks;
                bf.u[vv] = *(const unsigned int*)&Wlds[colBase + m][kb];
            }
            acc = __builtin_amdgcn_wmma_f32_16x16x32_bf16(
                      false, af.v, false, bf.v, (short)0, acc, false, false);
        }
        // D layout: VGPR r -> row mBase + r + 8g, col = colBase + lane%16.
        // Uniform branch: full tiles take the straight-line store path
        // (lanes 0-15 / 16-31 each cover a contiguous 64B row segment).
        float* yp = Y + (size_t)(mBase + 8 * g) * Ko + colBase + m;
        if (fullTile) {
            #pragma unroll
            for (int r = 0; r < 8; ++r) yp[(size_t)r * Ko] = acc[r];
        } else {
            #pragma unroll
            for (int r = 0; r < 8; ++r)
                if (mBase + r + 8 * g < n_rows) yp[(size_t)r * Ko] = acc[r];
        }
    }
}

// ---------------------------------------------------------------- attention scores
__global__ void k_att(const float* __restrict__ h, const float* __restrict__ att_src,
                      const float* __restrict__ att_dst, float* __restrict__ a_s,
                      float* __restrict__ a_d, int n)
{
    int i = blockIdx.x * blockDim.x + threadIdx.x;   // over n*H
    if (i >= n * 2) return;
    int node = i >> 1, hh = i & 1;
    const float* hp = h + node * 128 + hh * 64;
    float ss = 0.f, sd = 0.f;
    #pragma unroll 4
    for (int c = 0; c < 64; ++c) {
        float v = hp[c];
        ss += v * att_src[hh * 64 + c];
        sd += v * att_dst[hh * 64 + c];
    }
    a_s[i] = ss; a_d[i] = sd;
}

// ---------------------------------------------------------------- edge passes
__device__ __forceinline__ void edge_pair(const int* ei, int e, int E_, int& s, int& d) {
    if (e < E_) { s = ei[e]; d = ei[E_ + e]; }   // edge_index[0]=src row, [1]=dst row
    else        { s = d = e - E_; }              // self loops appended
}

__global__ void k_edge_max(const int* __restrict__ ei, const float* __restrict__ a_s,
                           const float* __restrict__ a_d, float* __restrict__ smax,
                           int E_, int Etot)
{
    int i = blockIdx.x * blockDim.x + threadIdx.x;
    if (i >= Etot * 2) return;
    int e = i >> 1, hh = i & 1;
    if (e + 4096 < E_) __builtin_prefetch(ei + e + 4096, 0, 1);
    int s, d; edge_pair(ei, e, E_, s, d);
    float l = a_s[s * 2 + hh] + a_d[d * 2 + hh];
    l = (l > 0.f) ? l : NEG_SLOPE * l;
    atomicMaxF(&smax[d * 2 + hh], l);
}

__global__ void k_edge_sum(const int* __restrict__ ei, const float* __restrict__ a_s,
                           const float* __restrict__ a_d, const float* __restrict__ smax,
                           float* __restrict__ ssum, int E_, int Etot)
{
    int i = blockIdx.x * blockDim.x + threadIdx.x;
    if (i >= Etot * 2) return;
    int e = i >> 1, hh = i & 1;
    if (e + 4096 < E_) __builtin_prefetch(ei + e + 4096, 0, 1);
    int s, d; edge_pair(ei, e, E_, s, d);
    float l = a_s[s * 2 + hh] + a_d[d * 2 + hh];
    l = (l > 0.f) ? l : NEG_SLOPE * l;
    atomicAdd(&ssum[d * 2 + hh], __expf(l - smax[d * 2 + hh]));
}

// one wave per edge; lane owns 4 consecutive channels of the 128-wide (H*C) row
__global__ __launch_bounds__(256)
void k_edge_agg(const int* __restrict__ ei, const float* __restrict__ h,
                const float* __restrict__ a_s, const float* __restrict__ a_d,
                const float* __restrict__ smax, const float* __restrict__ ssum,
                float* __restrict__ agg, int E_, int Etot)
{
    int tid = blockIdx.x * 256 + threadIdx.x;
    int e = tid >> 5;
    if (e >= Etot) return;
    int lane = tid & 31;
    if (e + 512 < E_) __builtin_prefetch(ei + e + 512, 0, 1);
    int s, d; edge_pair(ei, e, E_, s, d);
    int hh = lane >> 4;                      // channels 4*lane..4*lane+3 -> head lane/16
    float l = a_s[s * 2 + hh] + a_d[d * 2 + hh];
    l = (l > 0.f) ? l : NEG_SLOPE * l;
    float alpha = __expf(l - smax[d * 2 + hh]) / ssum[d * 2 + hh];
    const float4 hv = *(const float4*)(h + s * 128 + lane * 4);
    float* ap = agg + d * 128 + lane * 4;
    atomicAdd(ap + 0, hv.x * alpha);
    atomicAdd(ap + 1, hv.y * alpha);
    atomicAdd(ap + 2, hv.z * alpha);
    atomicAdd(ap + 3, hv.w * alpha);
}

// ---------------------------------------------------------------- head mean + bias
__global__ void k_head_mean(const float* __restrict__ agg, const float* __restrict__ bias,
                            float* __restrict__ out, int n)
{
    int i = blockIdx.x * blockDim.x + threadIdx.x;
    if (i >= n * 64) return;
    int node = i >> 6, c = i & 63;
    out[i] = 0.5f * (agg[node * 128 + c] + agg[node * 128 + 64 + c]) + bias[c];
}

// ---------------------------------------------------------------- BN statistics
__global__ __launch_bounds__(256)
void k_bn_stats(const float* __restrict__ x, float* __restrict__ sums /*[sum|sumsq]*/, int n)
{
    int c = threadIdx.x & 63;
    int grp = threadIdx.x >> 6;              // 4 row-groups per block
    float s = 0.f, s2 = 0.f;
    for (int r = blockIdx.x * 4 + grp; r < n; r += gridDim.x * 4) {
        float v = x[r * 64 + c];
        s += v; s2 += v * v;
    }
    __shared__ float ls[4][64], ls2[4][64];
    ls[grp][c] = s; ls2[grp][c] = s2;
    __syncthreads();
    if (grp == 0) {
        atomicAdd(&sums[c],      ls[0][c] + ls[1][c] + ls[2][c] + ls[3][c]);
        atomicAdd(&sums[64 + c], ls2[0][c] + ls2[1][c] + ls2[2][c] + ls2[3][c]);
    }
}

// ---------------------------------------------------------------- BN apply + skip/residual + GELU
__global__ void k_apply_skip(const float* __restrict__ xm, const float* __restrict__ sums,
                             const float* __restrict__ gamma, const float* __restrict__ beta,
                             const float* __restrict__ skip, const float* __restrict__ skipb,
                             float* __restrict__ out, int n)
{
    int i = blockIdx.x * blockDim.x + threadIdx.x;
    if (i >= n * 64) return;
    int c = i & 63;
    float inv_n = 1.0f / (float)n;
    float mu = sums[c] * inv_n;
    float var = sums[64 + c] * inv_n - mu * mu;
    float v = (xm[i] - mu) * rsqrtf(var + 1e-5f) * gamma[c] + beta[c];
    v += skip[i] + skipb[c];
    out[i] = gelu_tanh(v);
}

__global__ void k_apply_res(const float* __restrict__ xm, const float* __restrict__ sums,
                            const float* __restrict__ gamma, const float* __restrict__ beta,
                            const float* __restrict__ res, float* __restrict__ out, int n)
{
    int i = blockIdx.x * blockDim.x + threadIdx.x;
    if (i >= n * 64) return;
    int c = i & 63;
    float inv_n = 1.0f / (float)n;
    float mu = sums[c] * inv_n;
    float var = sums[64 + c] * inv_n - mu * mu;
    float v = (xm[i] - mu) * rsqrtf(var + 1e-5f) * gamma[c] + beta[c];
    v += res[i];
    out[i] = gelu_tanh(v);
}

// ---------------------------------------------------------------- global mean pool
__global__ void k_pool(const float* __restrict__ h2, const int* __restrict__ batch,
                       float* __restrict__ psum, float* __restrict__ pcnt, int n)
{
    int i = blockIdx.x * blockDim.x + threadIdx.x;
    if (i >= n * 64) return;
    int node = i >> 6, c = i & 63;
    int b = batch[node];
    atomicAdd(&psum[b * 64 + c], h2[i]);
    if (c == 0) atomicAdd(&pcnt[b], 1.0f);
}

__global__ void k_pool_final(const float* __restrict__ psum, const float* __restrict__ pcnt,
                             float* __restrict__ out)
{
    int i = blockIdx.x * blockDim.x + threadIdx.x;
    if (i >= 64 * 64) return;
    out[i] = psum[i] / fmaxf(pcnt[i >> 6], 1.0f);
}

// ================================================================ launch
extern "C" void kernel_launch(void* const* d_in, const int* in_sizes, int n_in,
                              void* d_out, int out_size, void* d_ws, size_t ws_size,
                              hipStream_t stream)
{
    const float* x        = (const float*)d_in[0];
    const int*   ei       = (const int*)d_in[1];
    const int*   batch    = (const int*)d_in[2];
    const float* W1       = (const float*)d_in[3];
    const float* att_s1   = (const float*)d_in[4];
    const float* att_d1   = (const float*)d_in[5];
    const float* bias1    = (const float*)d_in[6];
    const float* skipW    = (const float*)d_in[7];
    const float* skipb    = (const float*)d_in[8];
    const float* gamma1   = (const float*)d_in[9];
    const float* beta1    = (const float*)d_in[10];
    const float* W2       = (const float*)d_in[11];
    const float* att_s2   = (const float*)d_in[12];
    const float* att_d2   = (const float*)d_in[13];
    const float* bias2    = (const float*)d_in[14];
    const float* gamma2   = (const float*)d_in[15];
    const float* beta2    = (const float*)d_in[16];

    const int N    = in_sizes[2];        // batch_index length
    const int E_   = in_sizes[1] / 2;    // edge_index [2,E]
    const int Etot = E_ + N;             // + self loops

    // workspace carve-up (fp32)
    size_t off = 0;
    auto carve = [&](size_t nf) {
        float* p = (float*)((char*)d_ws + off);
        off += nf * sizeof(float);
        return p;
    };
    float* h_big = carve((size_t)N * 128);   // GEMM output / gather source
    float* agg   = carve((size_t)N * 128);   // scatter accumulator
    float* asrc  = carve((size_t)N * 2);
    float* adst  = carve((size_t)N * 2);
    float* smax  = carve((size_t)N * 2);
    float* ssum  = carve((size_t)N * 2);
    float* meanb = carve((size_t)N * 64);    // head-mean + bias
    float* buf64 = carve((size_t)N * 64);    // skip proj, then final layer-2 output
    float* hA    = carve((size_t)N * 64);    // post layer-1 node features
    float* bnsum = carve(128);               // [sum(64) | sumsq(64)]
    float* psum  = carve(64 * 64);
    float* pcnt  = carve(64);
    (void)ws_size; (void)n_in; (void)out_size;

    const int T = 256;
    const int gEH  = (Etot * 2 + T - 1) / T;           // per (edge, head)
    const int gEW  = (Etot * 32 + T - 1) / T;          // wave per edge
    const int gNC  = (N * 64 + T - 1) / T;             // per (node, channel)
    const int gNH  = (N * 2 + T - 1) / T;              // per (node, head)
    const int gGemm = (N + 15) / 16;

    // ---------------- layer 1 ----------------
    hipMemsetAsync(agg, 0, (size_t)N * 128 * sizeof(float), stream);
    hipMemsetAsync(ssum, 0, (size_t)N * 2 * sizeof(float), stream);
    hipMemsetAsync(bnsum, 0, 128 * sizeof(float), stream);
    k_fill<<<gNH, T, 0, stream>>>(smax, -INFINITY, N * 2);

    k_gemm_bf16<<<gGemm, 128, 0, stream>>>(x, W1, h_big, N, 128);
    k_gemm_bf16<<<gGemm, 128, 0, stream>>>(x, skipW, buf64, N, 64);
    k_att<<<gNH, T, 0, stream>>>(h_big, att_s1, att_d1, asrc, adst, N);

    k_edge_max<<<gEH, T, 0, stream>>>(ei, asrc, adst, smax, E_, Etot);
    k_edge_sum<<<gEH, T, 0, stream>>>(ei, asrc, adst, smax, ssum, E_, Etot);
    k_edge_agg<<<gEW, T, 0, stream>>>(ei, h_big, asrc, adst, smax, ssum, agg, E_, Etot);

    k_head_mean<<<gNC, T, 0, stream>>>(agg, bias1, meanb, N);
    k_bn_stats<<<256, T, 0, stream>>>(meanb, bnsum, N);
    k_apply_skip<<<gNC, T, 0, stream>>>(meanb, bnsum, gamma1, beta1, buf64, skipb, hA, N);

    // ---------------- layer 2 ----------------
    hipMemsetAsync(agg, 0, (size_t)N * 128 * sizeof(float), stream);
    hipMemsetAsync(ssum, 0, (size_t)N * 2 * sizeof(float), stream);
    hipMemsetAsync(bnsum, 0, 128 * sizeof(float), stream);
    k_fill<<<gNH, T, 0, stream>>>(smax, -INFINITY, N * 2);

    k_gemm_bf16<<<gGemm, 128, 0, stream>>>(hA, W2, h_big, N, 128);
    k_att<<<gNH, T, 0, stream>>>(h_big, att_s2, att_d2, asrc, adst, N);

    k_edge_max<<<gEH, T, 0, stream>>>(ei, asrc, adst, smax, E_, Etot);
    k_edge_sum<<<gEH, T, 0, stream>>>(ei, asrc, adst, smax, ssum, E_, Etot);
    k_edge_agg<<<gEW, T, 0, stream>>>(ei, h_big, asrc, adst, smax, ssum, agg, E_, Etot);

    k_head_mean<<<gNC, T, 0, stream>>>(agg, bias2, meanb, N);
    k_bn_stats<<<256, T, 0, stream>>>(meanb, bnsum, N);
    k_apply_res<<<gNC, T, 0, stream>>>(meanb, bnsum, gamma2, beta2, hA, buf64, N);

    // ---------------- pool ----------------
    hipMemsetAsync(psum, 0, (64 * 64 + 64) * sizeof(float), stream);
    k_pool<<<gNC, T, 0, stream>>>(buf64, batch, psum, pcnt, N);
    k_pool_final<<<(64 * 64 + T - 1) / T, T, 0, stream>>>(psum, pcnt, (float*)d_out);
}
